// MultiHeadLatentAttention_41936060678346
// MI455X (gfx1250) — matrix operand order repaired and checked by reading, not compiled, not run on hardware
//
#include <hip/hip_runtime.h>

#define NTHREADS 256

typedef __attribute__((ext_vector_type(16))) __bf16        v16bf;
typedef __attribute__((ext_vector_type(8)))  float         v8f;
typedef __attribute__((ext_vector_type(4)))  unsigned int  v4u;
typedef int v4i_g __attribute__((vector_size(16)));   // gcc-style, matches builtin proto

union FragU { v4u u[2]; v16bf bf; };

static constexpr int B_C       = 2;
static constexpr int S_C       = 2048;
static constexpr int HID_C     = 2048;
static constexpr int NH_C      = 16;
static constexpr int NKV_C     = 4;
static constexpr int HD_C      = 128;
static constexpr int KV_RANK_C = 512;
static constexpr int MROWS_C   = B_C * S_C;           // 4096
#define ATT_SCALE 0.088388347648318447f               // 1/sqrt(128)

// --------------------------------------------------------------- async copy
// gfx1250 memory->LDS DMA (ASYNCcnt-tracked). Guarded: falls back to VGPR
// staging if the builtins are not declared by this toolchain.
#if __has_builtin(__builtin_amdgcn_global_load_async_to_lds_b128) && \
    __has_builtin(__builtin_amdgcn_s_wait_asynccnt)
#define HAVE_ASYNC 1
#else
#define HAVE_ASYNC 0
#endif

__device__ __forceinline__ void cp16_g2l(const unsigned short* g,
                                         unsigned short* l) {
#if HAVE_ASYNC
  // generic LDS pointer: low 32 bits are the LDS offset (ISA flat->LDS rule)
  __builtin_amdgcn_global_load_async_to_lds_b128(
      (__attribute__((address_space(1))) v4i_g*)(size_t)g,
      (__attribute__((address_space(3))) v4i_g*)(unsigned int)(size_t)l,
      0, 0);
#else
  *(v4u*)l = *(const v4u*)g;
#endif
}

__device__ __forceinline__ void cp_wait() {
#if HAVE_ASYNC
  __builtin_amdgcn_s_wait_asynccnt(0);
#endif
}

// --------------------------------------------------------------- helpers
__device__ __forceinline__ unsigned short f2bf(float f) {
  unsigned int u = __float_as_uint(f);
  u += 0x7FFFu + ((u >> 16) & 1u);                    // round-to-nearest-even
  return (unsigned short)(u >> 16);
}

// A-matrix 16x32 bf16 fragment (ISA table): lane<16 -> row=lane, K {0..7,16..23};
// lane>=16 -> row=lane-16, K {8..15,24..31}.  Two b128 loads per lane.
__device__ __forceinline__ v16bf frag_a(const unsigned short* base, int stride) {
  const int lane = threadIdx.x & 31;
  const unsigned short* p = base + (lane & 15) * stride + ((lane < 16) ? 0 : 8);
  FragU f;
  f.u[0] = *(const v4u*)(p);
  f.u[1] = *(const v4u*)(p + 16);
  return f.bf;
}

// B-matrix 32x16 bf16 fragment, loaded from B^T stored row-major (N x K):
// lane<16 -> col N=lane, K {0..7,16..23}; lane>=16 -> col N=lane-16, K {8..15,24..31}.
__device__ __forceinline__ v16bf frag_b(const unsigned short* baseT, int stride) {
  const int lane = threadIdx.x & 31;
  const unsigned short* p = baseT + (lane & 15) * stride + ((lane < 16) ? 0 : 8);
  FragU f;
  f.u[0] = *(const v4u*)(p);
  f.u[1] = *(const v4u*)(p + 16);
  return f.bf;
}

__device__ __forceinline__ v8f wmma_bf16(v16bf a, v16bf b, v8f c) {
  return __builtin_amdgcn_wmma_f32_16x16x32_bf16(false, a, false, b, (short)0, c,
                                                 false, false);
}

// --------------------------------------------------------------- conversions
__global__ void cvt_f32_bf16(const float* __restrict__ src,
                             unsigned short* __restrict__ dst, int n) {
  for (int i = blockIdx.x * blockDim.x + threadIdx.x; i < n;
       i += gridDim.x * blockDim.x)
    dst[i] = f2bf(src[i]);
}

// src (R x C) f32 row-major -> dst (C x R) bf16 row-major, 32x32 LDS tiles
__global__ __launch_bounds__(NTHREADS)
void cvt_T_f32_bf16(const float* __restrict__ src,
                    unsigned short* __restrict__ dst, int R, int C) {
  __shared__ float tile[32][33];
  const int c0 = blockIdx.x * 32;
  const int r0 = blockIdx.y * 32;
  const int tx = threadIdx.x & 31;
  const int ty = threadIdx.x >> 5;  // 0..7
#pragma unroll
  for (int i = 0; i < 32; i += 8)
    tile[ty + i][tx] = src[(size_t)(r0 + ty + i) * C + c0 + tx];
  __syncthreads();
#pragma unroll
  for (int i = 0; i < 32; i += 8)
    dst[(size_t)(c0 + ty + i) * R + r0 + tx] = f2bf(tile[tx][ty + i]);
}

// --------------------------------------------------------------- GEMM
// C[M,N] = A[M,K] * B[K,N] with B supplied PRE-TRANSPOSED (BT is N x K).
// Block tile 128x256, 8 waves (2x4), wave tile 64x64, K-step 32.
// Double-buffered LDS; tiles staged with async memory->LDS DMA.
template <int OUTF32>
__global__ __launch_bounds__(NTHREADS)
void gemm_bf16_wmma(const unsigned short* __restrict__ A, int lda,
                    const unsigned short* __restrict__ BT, int ldbt,
                    void* __restrict__ Cv, int ldc, int M, int N, int K) {
  __shared__ __align__(16) unsigned short ldsA[2][128 * 40];   // [m][k]
  __shared__ __align__(16) unsigned short ldsBT[2][256 * 40];  // [n][k]

  const int tid  = threadIdx.x;
  const int lane = tid & 31;
  const int wave = tid >> 5;
  const int wm   = wave & 1;   // rows wm*64
  const int wn   = wave >> 1;  // cols wn*64 (0..3)
  const int blockRow = blockIdx.y * 128;
  const int blockCol = blockIdx.x * 256;

  v8f acc[4][4];
#pragma unroll
  for (int i = 0; i < 4; ++i)
#pragma unroll
    for (int j = 0; j < 4; ++j)
#pragma unroll
      for (int r = 0; r < 8; ++r) acc[i][j][r] = 0.0f;

  auto stage = [&](int k0, int buf) {
    // A tile 128x32 : 512 16B chunks, 2 per thread
#pragma unroll
    for (int i = 0; i < 2; ++i) {
      int c  = tid + i * 256;
      int r  = c >> 2;
      int k8 = (c & 3) * 8;
      cp16_g2l(&A[(size_t)(blockRow + r) * lda + k0 + k8],
               &ldsA[buf][r * 40 + k8]);
    }
    // BT tile 256x32 : 1024 16B chunks, 4 per thread
#pragma unroll
    for (int i = 0; i < 4; ++i) {
      int c  = tid + i * 256;
      int n  = c >> 2;
      int k8 = (c & 3) * 8;
      cp16_g2l(&BT[(size_t)(blockCol + n) * ldbt + k0 + k8],
               &ldsBT[buf][n * 40 + k8]);
    }
  };

  stage(0, 0);
  cp_wait();
  __syncthreads();

  for (int k0 = 0; k0 < K; k0 += 32) {
    const int buf = (k0 >> 5) & 1;
    if (k0 + 32 < K) stage(k0 + 32, buf ^ 1);  // overlap with WMMAs below

    v16bf af[4], bfr[4];
#pragma unroll
    for (int t = 0; t < 4; ++t)
      af[t] = frag_a(&ldsA[buf][(wm * 64 + t * 16) * 40], 40);
#pragma unroll
    for (int t = 0; t < 4; ++t)
      bfr[t] = frag_b(&ldsBT[buf][(wn * 64 + t * 16) * 40], 40);

#pragma unroll
    for (int mt = 0; mt < 4; ++mt)
#pragma unroll
      for (int nt = 0; nt < 4; ++nt)
        acc[mt][nt] = wmma_bf16(af[mt], bfr[nt], acc[mt][nt]);

    cp_wait();
    __syncthreads();
  }

  // epilogue: C/D layout -> lane holds col (lane&15), rows r + (lane<16?0:8)
  const int rowOff = (lane < 16) ? 0 : 8;
  const int colIdx = lane & 15;
#pragma unroll
  for (int mt = 0; mt < 4; ++mt)
#pragma unroll
    for (int nt = 0; nt < 4; ++nt) {
      int rbase = blockRow + wm * 64 + mt * 16 + rowOff;
      int cidx  = blockCol + wn * 64 + nt * 16 + colIdx;
#pragma unroll
      for (int r = 0; r < 8; ++r) {
        if (OUTF32)
          ((float*)Cv)[(size_t)(rbase + r) * ldc + cidx] = acc[mt][nt][r];
        else
          ((unsigned short*)Cv)[(size_t)(rbase + r) * ldc + cidx] =
              f2bf(acc[mt][nt][r]);
      }
    }
}

// --------------------------------------------------------------- flash attention
// Grid: (B*NH, S/128). 8 waves, each owns 16 query rows. Key blocks of 32
// streamed through LDS; scores never touch HBM. Causal mask applied analytically.
__global__ __launch_bounds__(NTHREADS)
void mla_flash_attn_kernel(const unsigned short* __restrict__ Q,   // (B*S, 2048)
                           const unsigned short* __restrict__ Km,  // (B*S, 512)
                           const unsigned short* __restrict__ KVm, // (B*S, 1024), v = cols 512..
                           unsigned short* __restrict__ Om) {      // (B*S, 2048)
  __shared__ __align__(16) unsigned short ldsK[32 * 136];   // [key][d]
  __shared__ __align__(16) unsigned short ldsVT[128 * 40];  // [d][key]
  __shared__ __align__(16) unsigned short ldsP[128 * 32];   // per-wave 16x32 P

  const int tid    = threadIdx.x;
  const int lane   = tid & 31;
  const int wave   = tid >> 5;
  const int bh     = blockIdx.x;
  const int b      = bh >> 4;   // / NH
  const int h      = bh & 15;
  const int kvh    = h >> 2;    // GQA: h / n_rep
  const int qBase  = blockIdx.y * 128;
  const int rowOff = (lane < 16) ? 0 : 8;
  const int col    = lane & 15;

  // Q fragments resident in registers: 16 rows x 128 d = 4 A-fragments
  v16bf qf[4];
  const unsigned short* qptr =
      Q + ((size_t)(b * S_C + qBase + wave * 16)) * (NH_C * HD_C) + h * HD_C;
#pragma unroll
  for (int t = 0; t < 4; ++t) qf[t] = frag_a(qptr + t * 32, NH_C * HD_C);

  v8f o[8];
  float m_s[8], l_s[8];
#pragma unroll
  for (int t = 0; t < 8; ++t)
#pragma unroll
    for (int r = 0; r < 8; ++r) o[t][r] = 0.0f;
#pragma unroll
  for (int r = 0; r < 8; ++r) { m_s[r] = -1e30f; l_s[r] = 0.0f; }

  const int nkb = (qBase >> 5) + 4;  // causal: keys up to qBase+127
  for (int kb = 0; kb < nkb; ++kb) {
    const int kStart = kb * 32;
    __syncthreads();  // previous block's LDS reads done
    // stage K block 32x128 via async DMA: 512 16B chunks, 2 per thread
#pragma unroll
    for (int i = 0; i < 2; ++i) {
      int c   = tid + i * 256;
      int key = c >> 4;
      int d8  = (c & 15) * 8;
      cp16_g2l(&Km[(size_t)(b * S_C + kStart + key) * (NKV_C * HD_C) +
                   kvh * HD_C + d8],
               &ldsK[key * 136 + d8]);
    }
    // stage V block transposed: ldsVT[d][key] (element transpose, manual)
#pragma unroll
    for (int i = 0; i < 16; ++i) {
      int c   = tid + i * 256;
      int key = c >> 7;
      int dd  = c & 127;
      ldsVT[dd * 40 + key] =
          KVm[(size_t)(b * S_C + kStart + key) * (KV_RANK_C + NKV_C * HD_C) +
              KV_RANK_C + kvh * HD_C + dd];
    }
    cp_wait();
    __syncthreads();

    // S = Q @ K^T : two 16x16 f32 tiles (keys 0..15, 16..31)
    v8f s0, s1;
#pragma unroll
    for (int r = 0; r < 8; ++r) { s0[r] = 0.0f; s1[r] = 0.0f; }
#pragma unroll
    for (int t = 0; t < 4; ++t) {
      v16bf k0f = frag_b(ldsK + 0 * 136 + t * 32, 136);
      v16bf k1f = frag_b(ldsK + 16 * 136 + t * 32, 136);
      s0 = wmma_bf16(qf[t], k0f, s0);
      s1 = wmma_bf16(qf[t], k1f, s1);
    }

    // online softmax (row stats live in the 16-lane half owning the row)
#pragma unroll
    for (int r = 0; r < 8; ++r) {
      const int qrow = qBase + wave * 16 + rowOff + r;
      float v0 = s0[r] * ATT_SCALE + ((kStart + col) > qrow ? -1e9f : 0.0f);
      float v1 = s1[r] * ATT_SCALE + ((kStart + 16 + col) > qrow ? -1e9f : 0.0f);
      float mx = fmaxf(v0, v1);
      mx = fmaxf(mx, __shfl_xor(mx, 1, 32));
      mx = fmaxf(mx, __shfl_xor(mx, 2, 32));
      mx = fmaxf(mx, __shfl_xor(mx, 4, 32));
      mx = fmaxf(mx, __shfl_xor(mx, 8, 32));
      float mnew  = fmaxf(m_s[r], mx);
      float alpha = __expf(m_s[r] - mnew);
      float p0 = __expf(v0 - mnew);
      float p1 = __expf(v1 - mnew);
      float rs = p0 + p1;
      rs += __shfl_xor(rs, 1, 32);
      rs += __shfl_xor(rs, 2, 32);
      rs += __shfl_xor(rs, 4, 32);
      rs += __shfl_xor(rs, 8, 32);
      l_s[r] = l_s[r] * alpha + rs;
      m_s[r] = mnew;
#pragma unroll
      for (int t = 0; t < 8; ++t) o[t][r] *= alpha;
      // P tile -> per-wave LDS scratch (C-layout -> row-major 16x32)
      ldsP[(wave * 16 + rowOff + r) * 32 + col]      = f2bf(p0);
      ldsP[(wave * 16 + rowOff + r) * 32 + 16 + col] = f2bf(p1);
    }

    // O += P @ V : P as A-fragment, V^T rows as B-fragments
    v16bf pf = frag_a(ldsP + wave * 16 * 32, 32);
#pragma unroll
    for (int t = 0; t < 8; ++t) {
      v16bf vtf = frag_b(ldsVT + (t * 16) * 40, 40);
      o[t] = wmma_bf16(pf, vtf, o[t]);
    }
  }

  // normalize + store bf16 attention output
#pragma unroll
  for (int t = 0; t < 8; ++t)
#pragma unroll
    for (int r = 0; r < 8; ++r) {
      float val = o[t][r] / l_s[r];
      int srow  = qBase + wave * 16 + rowOff + r;
      Om[(size_t)(b * S_C + srow) * (NH_C * HD_C) + h * HD_C + t * 16 + col] =
          f2bf(val);
    }
}

// --------------------------------------------------------------- launcher
extern "C" void kernel_launch(void* const* d_in, const int* in_sizes, int n_in,
                              void* d_out, int out_size, void* d_ws,
                              size_t ws_size, hipStream_t stream) {
  (void)in_sizes; (void)n_in; (void)out_size; (void)ws_size;
  const float* hidden = (const float*)d_in[0];
  // d_in[1] = attention_mask : unused, causal mask applied analytically
  const float* W_q    = (const float*)d_in[2];
  const float* W_kva  = (const float*)d_in[3];
  const float* W_kvb  = (const float*)d_in[4];
  const float* W_o    = (const float*)d_in[5];
  float* out          = (float*)d_out;

  char* ws = (char*)d_ws;
  size_t off = 0;
  auto alloc = [&](size_t bytes) -> void* {
    void* p = ws + off;
    off = (off + bytes + 255) & ~(size_t)255;
    return p;
  };

  unsigned short* hB    = (unsigned short*)alloc((size_t)MROWS_C * HID_C * 2);
  unsigned short* WqT   = (unsigned short*)alloc((size_t)HID_C * 2048 * 2);
  unsigned short* WkvaT = (unsigned short*)alloc((size_t)HID_C * 1024 * 2);
  unsigned short* WkvbT = (unsigned short*)alloc((size_t)512 * 512 * 2);
  unsigned short* WoT   = (unsigned short*)alloc((size_t)2048 * 2048 * 2);
  unsigned short* Qb    = (unsigned short*)alloc((size_t)MROWS_C * 2048 * 2);
  unsigned short* KVb   = (unsigned short*)alloc((size_t)MROWS_C * 1024 * 2);
  unsigned short* Kb    = (unsigned short*)alloc((size_t)MROWS_C * 512 * 2);
  unsigned short* Ab    = (unsigned short*)alloc((size_t)MROWS_C * 2048 * 2);

  // activations: fp32 -> bf16 ; weights: fp32 -> bf16 TRANSPOSED (N x K)
  cvt_f32_bf16<<<1024, NTHREADS, 0, stream>>>(hidden, hB, MROWS_C * HID_C);
  cvt_T_f32_bf16<<<dim3(2048 / 32, 2048 / 32), NTHREADS, 0, stream>>>(
      W_q, WqT, 2048, 2048);
  cvt_T_f32_bf16<<<dim3(1024 / 32, 2048 / 32), NTHREADS, 0, stream>>>(
      W_kva, WkvaT, 2048, 1024);
  cvt_T_f32_bf16<<<dim3(512 / 32, 512 / 32), NTHREADS, 0, stream>>>(
      W_kvb, WkvbT, 512, 512);
  cvt_T_f32_bf16<<<dim3(2048 / 32, 2048 / 32), NTHREADS, 0, stream>>>(
      W_o, WoT, 2048, 2048);

  // q = hidden @ W_q   (4096x2048 @ 2048x2048)
  gemm_bf16_wmma<0><<<dim3(2048 / 256, MROWS_C / 128), NTHREADS, 0, stream>>>(
      hB, HID_C, WqT, 2048, Qb, 2048, MROWS_C, 2048, HID_C);
  // kv = hidden @ W_kva (4096x2048 @ 2048x1024)
  gemm_bf16_wmma<0><<<dim3(1024 / 256, MROWS_C / 128), NTHREADS, 0, stream>>>(
      hB, HID_C, WkvaT, 2048, KVb, 1024, MROWS_C, 1024, HID_C);
  // k = compressed_kv @ W_kvb (4096x512 @ 512x512), A = KVb cols 0..511 (lda=1024)
  gemm_bf16_wmma<0><<<dim3(512 / 256, MROWS_C / 128), NTHREADS, 0, stream>>>(
      KVb, 1024, WkvbT, 512, Kb, 512, MROWS_C, 512, KV_RANK_C);

  // flash attention
  mla_flash_attn_kernel<<<dim3(B_C * NH_C, S_C / 128), NTHREADS, 0, stream>>>(
      Qb, Kb, KVb, Ab);

  // out = attn @ W_o  (4096x2048 @ 2048x2048) -> fp32
  gemm_bf16_wmma<1><<<dim3(2048 / 256, MROWS_C / 128), NTHREADS, 0, stream>>>(
      Ab, 2048, WoT, 2048, out, 2048, MROWS_C, 2048, 2048);
}
